// DenseGATConv_23124103922018
// MI455X (gfx1250) — matrix operand order repaired and checked by reading, not compiled, not run on hardware
//
#include <hip/hip_runtime.h>
#include <hip/hip_bf16.h>

#define NN 8192
#define CIN 256
#define COUT 128
#define NEG_SLOPE 0.2f
#define WST 40   // padded LDS row stride (bf16 elems): 80B rows -> conflict-free b128 reads

typedef __attribute__((ext_vector_type(16))) __bf16 v16bf;
typedef __attribute__((ext_vector_type(8)))  __bf16 v8bf;
typedef __attribute__((ext_vector_type(8)))  float  v8f;

__device__ __forceinline__ __bf16 f2bf(float f) {
    // round-to-nearest-even f32 -> bf16 via integer math
    unsigned u = __builtin_bit_cast(unsigned, f);
    unsigned r = u + 0x7FFFu + ((u >> 16) & 1u);
    unsigned short h = (unsigned short)(r >> 16);
    return __builtin_bit_cast(__bf16, h);
}

// ---------------------------------------------------------------------------
// Kernel 1: h = x @ W  (bf16 WMMA, f32 accum), write hT (bf16, [COUT][NN]),
// and a_src[m] = h[m,:]·att_src, a_dst[m] = h[m,:]·att_dst via shfl+atomics.
// ---------------------------------------------------------------------------
__global__ __launch_bounds__(256) void gat_h_kernel(
    const float* __restrict__ x, const float* __restrict__ W,
    const float* __restrict__ att_src, const float* __restrict__ att_dst,
    __bf16* __restrict__ hT, float* __restrict__ a_src, float* __restrict__ a_dst)
{
    __shared__ __bf16 xs[16 * CIN];       // 16 rows of x, bf16 (8 KB)

    const int m0 = blockIdx.x * 16;
    const int t  = threadIdx.x;

    for (int idx = t; idx < 16 * CIN; idx += 256) {
        const int r = idx >> 8;
        const int k = idx & (CIN - 1);
        xs[idx] = f2bf(x[(size_t)(m0 + r) * CIN + k]);
    }
    __syncthreads();

    const int lane = t & 31;
    const int wave = t >> 5;
    const int g    = lane >> 4;
    const int ln   = lane & 15;
    const int c    = wave * 16 + ln;      // output channel (B/N dim)

    v8f acc = {};
    for (int k0 = 0; k0 < CIN; k0 += 32) {
        v16bf a, b;
        const __bf16* xr = xs + ln * CIN + k0;   // A: M=ln, interleaved K
#pragma unroll
        for (int e = 0; e < 8; ++e) {
            a[e]     = xr[8 * g + e];
            a[8 + e] = xr[16 + 8 * g + e];
        }
        const float* wc = W + c;                 // B: N=c, interleaved K
#pragma unroll
        for (int e = 0; e < 16; ++e)
            b[e] = f2bf(wc[(size_t)(k0 + 8 * g + (e & 7) + ((e >> 3) << 4)) * COUT]);

        acc = __builtin_amdgcn_wmma_f32_16x16x32_bf16(
                  false, a, false, b, (short)0, acc, false, false);
    }

    const float as = att_src[c];
    const float ad = att_dst[c];
#pragma unroll
    for (int v = 0; v < 8; ++v) {
        const int m = m0 + v + 8 * g;            // C/D: M = v + 8*(lane/16)
        hT[(size_t)c * NN + m] = f2bf(acc[v]);
        float vs = acc[v] * as;
        float vd = acc[v] * ad;
#pragma unroll
        for (int msk = 1; msk < 16; msk <<= 1) {
            vs += __shfl_xor(vs, msk, 32);
            vd += __shfl_xor(vd, msk, 32);
        }
        if (ln == 0) {
            atomicAdd(&a_src[m], vs);
            atomicAdd(&a_dst[m], vd);
        }
    }
}

// ---------------------------------------------------------------------------
// Kernel 2: fused single pass over adj, double-buffered staging.
// Block = 16 output nodes j. 8 waves = 8 c-tiles of h.
// Weight tile stored TRANSPOSED in LDS ([jj][ii], stride WST) so the A
// fragment is read with two ds_load_b128 per lane. The exp/lrelu/pack path
// runs only when the thread's float2 of adj has a nonzero (sparse skip).
// ---------------------------------------------------------------------------
__global__ __launch_bounds__(256) void gat_agg_kernel(
    const float* __restrict__ adj, const __bf16* __restrict__ hT,
    const float* __restrict__ a_src, const float* __restrict__ a_dst,
    const float* __restrict__ bias, float* __restrict__ out)
{
    __shared__ __bf16 wtile[2][16 * WST];   // double-buffered alpha^T tile
    __shared__ float  denom_sh[16];

    const int j0   = blockIdx.x * 16;
    const int t    = threadIdx.x;
    const int q    = t & 7;               // column-pair group: jj = 2q, 2q+1
    const int ii   = t >> 3;              // row within chunk, 0..31
    const int lane = t & 31;
    const int wave = t >> 5;
    const int g    = lane >> 4;
    const int ln   = lane & 15;
    const int c    = wave * 16 + ln;      // output channel (B/N dim)

    if (t < 16) denom_sh[t] = 0.f;

    const float dj0 = a_dst[j0 + 2 * q];
    const float dj1 = a_dst[j0 + 2 * q + 1];
    const int  diag_i0 = j0 & ~31;        // the one chunk holding this block's diagonal
    float dsum0 = 0.f, dsum1 = 0.f;
    v8f acc = {};

    const __bf16* hc = hT + (size_t)c * NN;   // contiguous in i

    // ---- staging helper: compute weights for chunk at i0c into buf ----
    auto stage = [&](int i0c, __bf16* buf) {
        const int i = i0c + ii;
        float2 av = *(const float2*)(adj + (size_t)i * NN + j0 + 2 * q);
        if (i0c == diag_i0) {             // uniform branch: self loops
            if (i == j0 + 2 * q)     av.x = 1.0f;
            if (i == j0 + 2 * q + 1) av.y = 1.0f;
        }
        __bf16 w0b = __builtin_bit_cast(__bf16, (unsigned short)0);
        __bf16 w1b = w0b;
        if ((av.x != 0.f) | (av.y != 0.f)) {   // sparse skip: ~99% of lanes idle
            const float s  = a_src[i];
            const float t0 = s + dj0;
            const float t1 = s + dj1;
            const float l0 = fmaxf(t0, 0.f) + NEG_SLOPE * fminf(t0, 0.f);
            const float l1 = fmaxf(t1, 0.f) + NEG_SLOPE * fminf(t1, 0.f);
            const float w0 = av.x * __expf(l0);
            const float w1 = av.y * __expf(l1);
            dsum0 += w0;
            dsum1 += w1;
            w0b = f2bf(w0);
            w1b = f2bf(w1);
        }
        buf[(2 * q) * WST + ii]     = w0b;
        buf[(2 * q + 1) * WST + ii] = w1b;
    };

    stage(0, &wtile[0][0]);
    __syncthreads();

    const int NCH = NN / 32;
    for (int k = 0; k < NCH; ++k) {
        const int cur = k & 1;
        const __bf16* wt = &wtile[cur][0];

        // A fragment: two 16B vector LDS reads (issued before staging stores
        // so the in-order DS queue doesn't stall them behind next-chunk work)
        v8bf lo = *(const v8bf*)(wt + ln * WST + 16 * g);
        v8bf hi = *(const v8bf*)(wt + ln * WST + 16 * g + 8);
        // B fragment: 16 consecutive bf16 of hT (K = 32k + 16g .. +15)
        v16bf b = *(const v16bf*)(hc + 32 * k + 16 * g);

        if (k + 1 < NCH) stage(32 * (k + 1), &wtile[1 - cur][0]);
        if (k + 2 < NCH)   // pull chunk k+2 of adj toward the caches
            __builtin_prefetch(adj + (size_t)(32 * (k + 2) + ii) * NN + j0 + 2 * q, 0, 0);

        v16bf a;
#pragma unroll
        for (int e = 0; e < 8; ++e) { a[e] = lo[e]; a[8 + e] = hi[e]; }

        acc = __builtin_amdgcn_wmma_f32_16x16x32_bf16(
                  false, a, false, b, (short)0, acc, false, false);
        __syncthreads();
    }

    // ---- fold private denominator partials (one-time) ----
    atomicAdd(&denom_sh[2 * q],     dsum0);
    atomicAdd(&denom_sh[2 * q + 1], dsum1);
    __syncthreads();

    const float bc = bias[c];
#pragma unroll
    for (int v = 0; v < 8; ++v) {
        const int j = j0 + v + 8 * g;     // C/D: M = v + 8*(lane/16)
        out[(size_t)j * COUT + c] = acc[v] / denom_sh[v + 8 * g] + bc;
    }
}

// ---------------------------------------------------------------------------
extern "C" void kernel_launch(void* const* d_in, const int* in_sizes, int n_in,
                              void* d_out, int out_size, void* d_ws, size_t ws_size,
                              hipStream_t stream) {
    const float* x       = (const float*)d_in[0];
    const float* adj     = (const float*)d_in[1];
    const float* W       = (const float*)d_in[2];
    const float* att_src = (const float*)d_in[3];
    const float* att_dst = (const float*)d_in[4];
    const float* bias    = (const float*)d_in[5];
    float* out = (float*)d_out;

    char* ws = (char*)d_ws;
    __bf16* hT    = (__bf16*)ws;                                   // 2 MB
    float*  a_src = (float*)(ws + (size_t)COUT * NN * sizeof(__bf16));
    float*  a_dst = a_src + NN;

    hipMemsetAsync(a_src, 0, 2 * NN * sizeof(float), stream);
    gat_h_kernel<<<NN / 16, 256, 0, stream>>>(x, W, att_src, att_dst,
                                              hT, a_src, a_dst);
    gat_agg_kernel<<<NN / 16, 256, 0, stream>>>(adj, hT, a_src, a_dst,
                                                bias, out);
}
